// EnsembleMLP_79482664780332
// MI455X (gfx1250) — matrix operand order repaired and checked by reading, not compiled, not run on hardware
//
#include <hip/hip_runtime.h>
#include <cstdint>

// EnsembleMLP fused kernel for MI455X (gfx1250, wave32, WMMA).
//   pred[b] = (1/M) * wacc . tanh(x[b] @ W_in.T),  wacc scatter-built from W_pred/ids.
// bf16-WMMA GEMM with tanh+GEMV epilogue (hidden never touches HBM).
// Double-buffered LDS (one barrier per K-slab), register-staged global loads
// one slab ahead, and a 2-deep B-fragment rotation so ds_loads overlap WMMAs.

typedef __attribute__((ext_vector_type(16))) __bf16 v16bf;
typedef __attribute__((ext_vector_type(8)))  __bf16 v8bf;
typedef __attribute__((ext_vector_type(2)))  __bf16 v2bf;
typedef __attribute__((ext_vector_type(8)))  float  v8f;

#define B_SZ   8192
#define IN_SZ  1024
#define HID_SZ 4096
#define M_SZ   64
#define D_SZ   256

__device__ __forceinline__ unsigned pk2(float a, float b) {
    v2bf p; p[0] = (__bf16)a; p[1] = (__bf16)b;   // v_cvt_pk_bf16_f32
    return __builtin_bit_cast(unsigned, p);
}

__global__ void zero_kernel(float* __restrict__ wacc, float* __restrict__ out) {
    int t = blockIdx.x * blockDim.x + threadIdx.x;
    if (t < HID_SZ) wacc[t] = 0.0f;
    int o = t - HID_SZ;
    if (o >= 0 && o < B_SZ) out[o] = 0.0f;
}

__global__ void scatter_kernel(const float* __restrict__ W_pred,
                               const int* __restrict__ ids,
                               float* __restrict__ wacc) {
    int t = blockIdx.x * blockDim.x + threadIdx.x;   // 0 .. M*D-1
    if (t < M_SZ * D_SZ)
        atomicAdd(&wacc[ids[t]], W_pred[t] * (1.0f / (float)M_SZ));
}

__global__ __launch_bounds__(256)
void fused_gemm_tanh_dot(const float* __restrict__ x,
                         const float* __restrict__ Win,
                         const float* __restrict__ wacc,
                         float* __restrict__ out) {
    // Double-buffered: 128 rows x 32-K slab of x + 128 hid-cols x 32-K slab of W,
    // bf16, row stride 40 halfwords (80 B) -> 16-B fragment loads conflict-free.
    __shared__ unsigned short lds_x[2][128][40];
    __shared__ unsigned short lds_w[2][128][40];

    const int t    = threadIdx.x;
    const int lane = t & 31;
    const int wv   = t >> 5;                 // wave 0..7, owns rows wv*16..wv*16+15
    const int rowBase = blockIdx.x * 128;
    const int hBase   = blockIdx.y * 128;

    // staging coords: each thread moves 16 floats of x and 16 of W per slab
    const int ldRow = t >> 1;                // 0..127
    const int ldK   = (t & 1) * 16;          // 0 or 16
    const float* xPtr = x   + (size_t)(rowBase + ldRow) * IN_SZ + ldK;
    const float* wPtr = Win + (size_t)(hBase   + ldRow) * IN_SZ + ldK;

    // WMMA bf16 fragment addressing (ISA 7.12.2 layouts)
    const int half = lane >> 4;              // 0: lanes 0-15, 1: lanes 16-31
    const int l16  = lane & 15;
    const int aRow = wv * 16 + l16;          // A: lane's matrix row
    const int aK   = half * 8;               // A: K chunks {aK..aK+7},{aK+16..aK+23}
    const int bK   = half * 8;               // B: K chunks {bK..bK+7},{bK+16..bK+23}

    v8f acc[8];
#pragma unroll
    for (int ct = 0; ct < 8; ++ct) acc[ct] = (v8f){0,0,0,0,0,0,0,0};

    float4 sx[4], sw[4];                     // one K-slab staged in registers

    // ---- prologue: slab 0 -> buf 0; issue slab-1 loads; one barrier ----
    {
        const float4* xs = (const float4*)xPtr;
        const float4* ws = (const float4*)wPtr;
#pragma unroll
        for (int v = 0; v < 4; ++v) { sx[v] = xs[v]; sw[v] = ws[v]; }
        unsigned* dx = (unsigned*)&lds_x[0][ldRow][ldK];
        unsigned* dw = (unsigned*)&lds_w[0][ldRow][ldK];
#pragma unroll
        for (int v = 0; v < 4; ++v) {
            dx[v * 2 + 0] = pk2(sx[v].x, sx[v].y);
            dx[v * 2 + 1] = pk2(sx[v].z, sx[v].w);
            dw[v * 2 + 0] = pk2(sw[v].x, sw[v].y);
            dw[v * 2 + 1] = pk2(sw[v].z, sw[v].w);
        }
        const float4* xs1 = (const float4*)(xPtr + 32);
        const float4* ws1 = (const float4*)(wPtr + 32);
#pragma unroll
        for (int v = 0; v < 4; ++v) { sx[v] = xs1[v]; sw[v] = ws1[v]; }
    }
    __syncthreads();

#pragma unroll 2
    for (int kb = 0, p = 0; kb < IN_SZ; kb += 32, p ^= 1) {
        // ---- compute slab k from buf p ----
        const unsigned short (*lx)[40] = lds_x[p];
        const unsigned short (*lw)[40] = lds_w[p];

        v8bf alo = *(const v8bf*)&lx[aRow][aK];
        v8bf ahi = *(const v8bf*)&lx[aRow][aK + 16];
        v16bf afrag = __builtin_shufflevector(alo, ahi,
            0,1,2,3,4,5,6,7,8,9,10,11,12,13,14,15);

        auto loadB = [&](int ct) -> v16bf {
            const int bCol = ct * 16 + l16;
            v8bf blo = *(const v8bf*)&lw[bCol][bK];
            v8bf bhi = *(const v8bf*)&lw[bCol][bK + 16];
            return __builtin_shufflevector(blo, bhi,
                0,1,2,3,4,5,6,7,8,9,10,11,12,13,14,15);
        };

        v16bf bcur = loadB(0);
#pragma unroll
        for (int ct = 0; ct < 8; ++ct) {
            v16bf bnext = bcur;
            if (ct < 7) bnext = loadB(ct + 1);          // in flight during WMMA
            acc[ct] = __builtin_amdgcn_wmma_f32_16x16x32_bf16(
                false, afrag, false, bcur, (short)0, acc[ct], false, false);
            bcur = bnext;
        }

        // ---- commit staged slab k+1 into buf 1-p (overlaps with WMMAs above) ----
        if (kb + 32 < IN_SZ) {
            unsigned* dx = (unsigned*)&lds_x[p ^ 1][ldRow][ldK];
            unsigned* dw = (unsigned*)&lds_w[p ^ 1][ldRow][ldK];
#pragma unroll
            for (int v = 0; v < 4; ++v) {
                dx[v * 2 + 0] = pk2(sx[v].x, sx[v].y);
                dx[v * 2 + 1] = pk2(sx[v].z, sx[v].w);
                dw[v * 2 + 0] = pk2(sw[v].x, sw[v].y);
                dw[v * 2 + 1] = pk2(sw[v].z, sw[v].w);
            }
        }
        // ---- issue global loads for slab k+2; a full slab of compute to land ----
        if (kb + 64 < IN_SZ) {
            const float4* xs = (const float4*)(xPtr + kb + 64);
            const float4* ws = (const float4*)(wPtr + kb + 64);
#pragma unroll
            for (int v = 0; v < 4; ++v) { sx[v] = xs[v]; sw[v] = ws[v]; }
            if (kb + 96 < IN_SZ) {
                __builtin_prefetch(xPtr + kb + 96, 0, 0);   // global_prefetch_b8
                __builtin_prefetch(wPtr + kb + 96, 0, 0);
            }
        }
        __syncthreads();                     // one barrier per K-slab
    }

    // ---- epilogue: tanh, dot with wacc, half-wave reduce, atomic accumulate ----
    // C layout: acc[ct][r] = hidden[rowBase + wv*16 + r + half*8, hBase + ct*16 + l16]
    float partial[8];
#pragma unroll
    for (int r = 0; r < 8; ++r) partial[r] = 0.0f;

#pragma unroll
    for (int ct = 0; ct < 8; ++ct) {
        float wc = wacc[hBase + ct * 16 + l16];
#pragma unroll
        for (int r = 0; r < 8; ++r)
            partial[r] += tanhf(acc[ct][r]) * wc;
    }

#pragma unroll
    for (int r = 0; r < 8; ++r) {
        partial[r] += __shfl_xor(partial[r], 1, 32);
        partial[r] += __shfl_xor(partial[r], 2, 32);
        partial[r] += __shfl_xor(partial[r], 4, 32);
        partial[r] += __shfl_xor(partial[r], 8, 32);
    }
    if (l16 == 0) {
#pragma unroll
        for (int r = 0; r < 8; ++r) {
            int row = rowBase + wv * 16 + half * 8 + r;
            atomicAdd(&out[row], partial[r]);
        }
    }
}

extern "C" void kernel_launch(void* const* d_in, const int* in_sizes, int n_in,
                              void* d_out, int out_size, void* d_ws, size_t ws_size,
                              hipStream_t stream) {
    const float* x      = (const float*)d_in[0];   // (B, IN)
    const float* W_in   = (const float*)d_in[1];   // (HID, IN)
    const float* W_pred = (const float*)d_in[2];   // (M, 1, D)
    const int*   ids    = (const int*)d_in[3];     // (M, D)
    float* out  = (float*)d_out;                   // (B, 1)
    float* wacc = (float*)d_ws;                    // HID floats of scratch

    (void)in_sizes; (void)n_in; (void)out_size; (void)ws_size;

    zero_kernel<<<(HID_SZ + B_SZ + 255) / 256, 256, 0, stream>>>(wacc, out);
    scatter_kernel<<<(M_SZ * D_SZ + 255) / 256, 256, 0, stream>>>(W_pred, ids, wacc);

    dim3 grid(B_SZ / 128, HID_SZ / 128);           // 64 x 32 blocks
    fused_gemm_tanh_dot<<<grid, 256, 0, stream>>>(x, W_in, wacc, out);
}
